// EdgeStateUpdate_35691178230144
// MI455X (gfx1250) — compile-verified, matrix operand order checked by reading, not run on hardware
//
#include <hip/hip_runtime.h>
#include <hip/hip_bf16.h>
#include <stdint.h>

// ---------------------------------------------------------------------------
// Edge-MLP: out = silu([s[snd] | s[rcv] | e_state | e_len] @ W1 + b1) @ W2 + b2
// E = 320000, NODE_DIM = EDGE_DIM = 256, IN_DIM = 769.
//
// MI455X (gfx1250, wave32) design:
//  * f16 tensor path: v_wmma_f32_16x16x32_f16. GEMM1 K = 768 = 24*32 (the
//    769th column, edge_len, is a rank-1 epilogue update); GEMM2 K = 256.
//  * W1/W2 pre-converted fp32->f16 into d_ws in WMMA-B fragment order so each
//    lane reads its whole v16h fragment with two coalesced global b128 loads.
//  * Block = 256 threads (8 waves) x 64 edges. Each wave: 2 N-tiles x 4
//    M-tiles = 8 accumulators; each B fragment in registers feeds 4 WMMAs
//    (4x less L2 B-traffic than an M=16 blocking).
//  * A tile gathered + converted to f16 in LDS (padded row stride 776/264 ->
//    4-dword row skew, low-conflict ds_load_b128 fragment reads).
//
// Workspace requirement: (24+8)*16*32*16 * 2 bytes = 512 KiB in d_ws.
// ---------------------------------------------------------------------------

typedef __attribute__((ext_vector_type(16))) _Float16 v16h;
typedef __attribute__((ext_vector_type(8)))  float    v8f;

struct Pack32 { uint4 lo; uint4 hi; };   // 32 bytes == one v16h fragment

// K-offset inside a 32-wide K tile for fragment element `idx` on lane `lane`
// (ISA 7.12.2, 16-bit A 16x32 layout; B 32x16 mirrors it with M->N):
//   lanes 0-15  : K = (idx&7) + (idx>=8 ? 16 : 0)
//   lanes 16-31 : K = 8 + (idx&7) + (idx>=8 ? 16 : 0)
__device__ __forceinline__ int kmap(int lane, int idx) {
    return ((lane & 16) ? 8 : 0) + (idx & 7) + ((idx & 8) ? 16 : 0);
}

// -------------------- weight pre-swizzle (fp32 -> f16 fragments) ----------
// dst[((kt*16 + nt)*32 + lane)*16 + idx] =
//     (f16) W[(kt*32 + kmap(lane,idx))*256 + nt*16 + lane%16]
__global__ void prep_weights_kernel(const float* __restrict__ W,
                                    _Float16* __restrict__ dst,
                                    int n_ktiles) {
    int gid = blockIdx.x * blockDim.x + threadIdx.x;
    int total = n_ktiles * 16 * 32 * 16;
    if (gid >= total) return;
    int idx  = gid & 15;
    int lane = (gid >> 4) & 31;
    int nt   = (gid >> 9) & 15;
    int kt   = gid >> 13;
    int k = kt * 32 + kmap(lane, idx);
    int n = nt * 16 + (lane & 15);
    dst[gid] = (_Float16)W[(size_t)k * 256 + n];
}

// ------------------------------- main kernel ------------------------------
__global__ __launch_bounds__(256) void edge_mlp_kernel(
    const float* __restrict__ scalars,     // [N_NODES, 256]
    const int*   __restrict__ edge_index,  // [2, E] (int32)
    const float* __restrict__ edge_len,    // [E]
    const float* __restrict__ edge_state,  // [E, 256]
    const float* __restrict__ W1,          // [769, 256] fp32 (row 768 used here)
    const float* __restrict__ b1,          // [256]
    const float* __restrict__ b2,          // [256]
    const _Float16* __restrict__ w1t,      // pre-swizzled W1[0:768,:] fragments
    const _Float16* __restrict__ w2t,      // pre-swizzled W2 fragments
    float* __restrict__ out,               // [E, 256]
    int E)
{
    // A tile: 64 edges x 768 f16, row stride 776 (1552 B: 16B-aligned, and
    // 388 dwords -> 4-bank row skew => conflict-light b128 fragment reads).
    __shared__ alignas(16) _Float16 Abuf[64][776];
    // h tile: 64 x 256 f16, stride 264 (same skew trick).
    __shared__ alignas(16) _Float16 Hbuf[64][264];
    __shared__ float elen_s[64];

    const int tid = threadIdx.x;
    const int e0  = blockIdx.x * 64;

    // ---- Phase 1: gather + fp32->f16 convert into LDS -------------------
    // 64 rows x 192 float4 chunks; 4 threads cooperate on one row.
    {
        const int row  = tid >> 2;        // 0..63
        const int sub  = tid & 3;         // 0..3
        int e = e0 + row; if (e >= E) e = E - 1;
        const int s  = edge_index[e];
        const int r  = edge_index[E + e];
        const float* srcA = scalars    + (size_t)s * 256;  // cols 0..255
        const float* srcB = scalars    + (size_t)r * 256;  // cols 256..511
        const float* srcC = edge_state + (size_t)e * 256;  // cols 512..767
        #pragma unroll
        for (int c = 0; c < 192; c += 4) {
            int c4  = c + sub;            // chunk index 0..191
            int col = c4 * 4;
            const float* src = (col < 256)   ? (srcA + col)
                             : (col < 512)   ? (srcB + col - 256)
                                             : (srcC + col - 512);
            float4 v = *(const float4*)src;
            union { _Float16 h[4]; uint2 u; } p;
            p.h[0] = (_Float16)v.x; p.h[1] = (_Float16)v.y;
            p.h[2] = (_Float16)v.z; p.h[3] = (_Float16)v.w;
            *(uint2*)&Abuf[row][col] = p.u;        // ds_store_b64
        }
        if (sub == 0) elen_s[row] = edge_len[e];
    }
    __syncthreads();

    const int lane = tid & 31;
    const int wave = tid >> 5;               // 0..7  -> N-tiles wave*2, wave*2+1
    const int halfofs = (lane & 16) ? 8 : 0; // K/M offset for upper half-wave
    const int mrow = lane & 15;

    v8f acc[4][2];
    #pragma unroll
    for (int m = 0; m < 4; ++m)
        #pragma unroll
        for (int t = 0; t < 2; ++t)
            acc[m][t] = (v8f){0,0,0,0,0,0,0,0};

    // ---- Phase 2: GEMM1, K = 768 = 24 * 32 ------------------------------
    for (int kt = 0; kt < 24; ++kt) {
        // two B fragments (N-tiles wave*2, wave*2+1), reused by 4 M-tiles
        v16h bv[2];
        #pragma unroll
        for (int t = 0; t < 2; ++t) {
            int nt = wave * 2 + t;
            const uint4* wp =
                (const uint4*)(w1t + ((size_t)(kt * 16 + nt) * 32 + lane) * 16);
            Pack32 bp; bp.lo = wp[0]; bp.hi = wp[1];   // global_load_b128 x2
            bv[t] = __builtin_bit_cast(v16h, bp);
        }
        const int koff = kt * 32 + halfofs;
        #pragma unroll
        for (int m = 0; m < 4; ++m) {
            Pack32 ap;
            ap.lo = *(const uint4*)&Abuf[m * 16 + mrow][koff];      // ds b128
            ap.hi = *(const uint4*)&Abuf[m * 16 + mrow][koff + 16];
            v16h av = __builtin_bit_cast(v16h, ap);
            acc[m][0] = __builtin_amdgcn_wmma_f32_16x16x32_f16(
                false, av, false, bv[0], (short)0, acc[m][0], false, false);
            acc[m][1] = __builtin_amdgcn_wmma_f32_16x16x32_f16(
                false, av, false, bv[1], (short)0, acc[m][1], false, false);
        }
    }

    // ---- Epilogue 1: + b1 + edge_len * W1[768,:], SiLU, h -> LDS (f16) --
    #pragma unroll
    for (int t = 0; t < 2; ++t) {
        int n = (wave * 2 + t) * 16 + mrow;
        float bias = b1[n];
        float wl   = W1[(size_t)768 * 256 + n];   // rank-1 column (L2-resident)
        #pragma unroll
        for (int m = 0; m < 4; ++m) {
            #pragma unroll
            for (int j = 0; j < 8; ++j) {
                int row = m * 16 + j + halfofs;    // C layout: VGPR j -> M=j(+8)
                float x = acc[m][t][j] + bias + elen_s[row] * wl;
                float h = x / (1.0f + __expf(-x)); // silu
                Hbuf[row][n] = (_Float16)h;
            }
        }
    }
    __syncthreads();

    #pragma unroll
    for (int m = 0; m < 4; ++m)
        #pragma unroll
        for (int t = 0; t < 2; ++t)
            acc[m][t] = (v8f){0,0,0,0,0,0,0,0};

    // ---- Phase 3: GEMM2, K = 256 = 8 * 32 -------------------------------
    for (int kt = 0; kt < 8; ++kt) {
        v16h bv[2];
        #pragma unroll
        for (int t = 0; t < 2; ++t) {
            int nt = wave * 2 + t;
            const uint4* wp =
                (const uint4*)(w2t + ((size_t)(kt * 16 + nt) * 32 + lane) * 16);
            Pack32 bp; bp.lo = wp[0]; bp.hi = wp[1];
            bv[t] = __builtin_bit_cast(v16h, bp);
        }
        const int koff = kt * 32 + halfofs;
        #pragma unroll
        for (int m = 0; m < 4; ++m) {
            Pack32 ap;
            ap.lo = *(const uint4*)&Hbuf[m * 16 + mrow][koff];
            ap.hi = *(const uint4*)&Hbuf[m * 16 + mrow][koff + 16];
            v16h av = __builtin_bit_cast(v16h, ap);
            acc[m][0] = __builtin_amdgcn_wmma_f32_16x16x32_f16(
                false, av, false, bv[0], (short)0, acc[m][0], false, false);
            acc[m][1] = __builtin_amdgcn_wmma_f32_16x16x32_f16(
                false, av, false, bv[1], (short)0, acc[m][1], false, false);
        }
    }

    // ---- Epilogue 2: + b2, store fp32 -----------------------------------
    #pragma unroll
    for (int t = 0; t < 2; ++t) {
        int n = (wave * 2 + t) * 16 + mrow;
        float bias = b2[n];
        #pragma unroll
        for (int m = 0; m < 4; ++m) {
            #pragma unroll
            for (int j = 0; j < 8; ++j) {
                int row = m * 16 + j + halfofs;
                int e = e0 + row;
                if (e < E) out[(size_t)e * 256 + n] = acc[m][t][j] + bias;
            }
        }
    }
}

// ------------------------------- launcher ---------------------------------
extern "C" void kernel_launch(void* const* d_in, const int* in_sizes, int n_in,
                              void* d_out, int out_size, void* d_ws, size_t ws_size,
                              hipStream_t stream) {
    const float* scalars    = (const float*)d_in[0];
    const int*   edge_index = (const int*)  d_in[1];
    const float* edge_len   = (const float*)d_in[2];
    const float* edge_state = (const float*)d_in[3];
    const float* W1         = (const float*)d_in[4];
    const float* b1         = (const float*)d_in[5];
    const float* W2         = (const float*)d_in[6];
    const float* b2         = (const float*)d_in[7];
    float* out = (float*)d_out;

    const int E = in_sizes[2];                 // edge_len element count

    // Workspace layout: [w1 tiles | w2 tiles], f16.
    const int W1T_ELEMS = 24 * 16 * 32 * 16;   // 196608
    const int W2T_ELEMS =  8 * 16 * 32 * 16;   //  65536
    _Float16* w1t = (_Float16*)d_ws;
    _Float16* w2t = w1t + W1T_ELEMS;

    prep_weights_kernel<<<(W1T_ELEMS + 255) / 256, 256, 0, stream>>>(W1, w1t, 24);
    prep_weights_kernel<<<(W2T_ELEMS + 255) / 256, 256, 0, stream>>>(W2, w2t, 8);

    int nblk = (E + 63) / 64;                  // 5000 blocks of 256 threads
    edge_mlp_kernel<<<nblk, 256, 0, stream>>>(
        scalars, edge_index, edge_len, edge_state,
        W1, b1, b2, w1t, w2t, out, E);
}
